// LayerNormDenseGeneral_3564822855851
// MI455X (gfx1250) — matrix operand order repaired and checked by reading, not compiled, not run on hardware
//
#include <hip/hip_runtime.h>
#include <stdint.h>

// ---- problem constants (match the reference) ----
#define SEQ      4096
#define BATCH    8
#define HIDDEN   1024
#define FEATURES 1024
#define ROWS     (SEQ*BATCH)        // 32768 GEMM rows
#define Z_ELEMS  (ROWS*FEATURES)    // 33554432 floats (first output)
#define EPSF     1e-6f

typedef __attribute__((ext_vector_type(16))) __bf16        v16bf;
typedef __attribute__((ext_vector_type(8)))  float         v8f;
typedef __attribute__((ext_vector_type(8)))  unsigned int  v8u;

// f32 -> bf16 with round-to-nearest-even (bit twiddle; avoids relying on
// __bf16 conversion codegen details).
__device__ __forceinline__ unsigned int f2bf(float f) {
    unsigned int u = __float_as_uint(f);
    u += 0x7FFFu + ((u >> 16) & 1u);
    return u >> 16;
}

// Assemble a 16-half bf16 WMMA operand from two 16-byte chunks.
__device__ __forceinline__ v16bf packbf(uint4 lo, uint4 hi) {
    v8u v;
    v[0] = lo.x; v[1] = lo.y; v[2] = lo.z; v[3] = lo.w;
    v[4] = hi.x; v[5] = hi.y; v[6] = hi.z; v[7] = hi.w;
    return __builtin_bit_cast(v16bf, v);
}

// -------------------------------------------------------------------------
// Kernel 1: swizzle the f32 weight matrix W[K=1024][N=1024] (row major) into
// bf16 WMMA-B fragments in workspace.
// Layout: linear half index = ((ntile*32 + ktile)*32 + lane)*16 + h
//   value = W[ ktile*32 + ((lane&16)?8:0) + (h<8 ? h : h+8) ][ ntile*16 + (lane&15) ]
// Matches V_WMMA_F32_16X16X32_BF16's B (32x16) lane layout: the GEMM loads
// each fragment as two fully-coalesced b128 loads per lane.
// -------------------------------------------------------------------------
__global__ __launch_bounds__(256)
void prep_b_kernel(const float* __restrict__ w, unsigned short* __restrict__ bswz) {
    int idx   = blockIdx.x * 256 + threadIdx.x;        // 0 .. 1M-1
    int h     = idx & 15;
    int lane  = (idx >> 4) & 31;
    int ktile = (idx >> 9) & 31;
    int ntile = idx >> 14;
    int k = ktile * 32 + ((lane & 16) ? 8 : 0) + (h < 8 ? h : h + 8);
    int n = ntile * 16 + (lane & 15);
    bswz[idx] = (unsigned short)f2bf(w[k * FEATURES + n]);
}

// -------------------------------------------------------------------------
// Kernel 2: LayerNorm (scale-only affine) over the hidden axis.
// One 256-thread block per row (4 elements/thread), LDS tree reduction.
// Writes the f32 LN output (second output, also the GEMM A operand).
// -------------------------------------------------------------------------
__global__ __launch_bounds__(256)
void ln_kernel(const float* __restrict__ x, const float* __restrict__ scale,
               float* __restrict__ ln) {
    __shared__ float s_sum[256];
    __shared__ float s_sq[256];
    const int row = blockIdx.x;
    const int t   = threadIdx.x;
    const float* xr = x + (size_t)row * HIDDEN;

    float v[4];
    float sum = 0.f, sq = 0.f;
#pragma unroll
    for (int j = 0; j < 4; ++j) {
        v[j] = xr[t + 256 * j];
        sum += v[j];
        sq  += v[j] * v[j];
    }
    s_sum[t] = sum; s_sq[t] = sq;
    __syncthreads();
#pragma unroll
    for (int off = 128; off > 0; off >>= 1) {
        if (t < off) { s_sum[t] += s_sum[t + off]; s_sq[t] += s_sq[t + off]; }
        __syncthreads();
    }
    const float mean = s_sum[0] * (1.0f / HIDDEN);
    const float var  = s_sq[0]  * (1.0f / HIDDEN) - mean * mean;
    const float rstd = rsqrtf(var + EPSF);

    float* lr = ln + (size_t)row * HIDDEN;
#pragma unroll
    for (int j = 0; j < 4; ++j) {
        int h = t + 256 * j;
        lr[h] = (v[j] - mean) * rstd * scale[h];
    }
}

// -------------------------------------------------------------------------
// Kernel 3: WMMA GEMM  z[32768x1024] = lnA[32768x1024] * B[1024x1024]
//   grid.x = 1024 blocks, each owning a 32-row M-block; block = 8 waves.
//   Block stages 32x1024 of A as bf16 in LDS (64 KB) once.
//   Each wave owns a 128-column slab and BOTH 16-row M-tiles:
//   16 f32 accumulator tiles (128 VGPRs), K loop of 32 steps,
//   16 v_wmma_f32_16x16x32_bf16 per step (512 total per wave).
//   Every B fragment (2x global b128) feeds 2 WMMAs; next K-step's B slab
//   is pulled near via global_prefetch while the current WMMAs execute.
// -------------------------------------------------------------------------
__global__ __launch_bounds__(256)
void gemm_kernel(const float* __restrict__ ln,
                 const unsigned short* __restrict__ bswz,
                 float* __restrict__ z) {
    __shared__ __align__(16) unsigned short smemA[32 * HIDDEN]; // 64 KB bf16

    const int t    = threadIdx.x;
    const int mblk = blockIdx.x;                 // 32-row block index

    // ---- stage A: 32 rows x 1024 f32 -> bf16 in LDS (coalesced b128 reads)
    const float4* a4 = reinterpret_cast<const float4*>(ln + (size_t)mblk * 32 * HIDDEN);
    unsigned int* s32 = reinterpret_cast<unsigned int*>(smemA);
#pragma unroll
    for (int j = 0; j < 32; ++j) {               // 8192 float4 / 256 threads
        int i = t + 256 * j;
        float4 f = a4[i];
        s32[i * 2 + 0] = f2bf(f.x) | (f2bf(f.y) << 16);
        s32[i * 2 + 1] = f2bf(f.z) | (f2bf(f.w) << 16);
    }
    __syncthreads();

    const int lane  = t & 31;
    const int wave  = t >> 5;
    const int nbase = wave * 128;                // this wave's column slab
    const int mrow  = lane & 15;
    const int hi    = (lane & 16) ? 8 : 0;       // A K-offset AND C row-offset

    v8f acc0[8], acc1[8];                        // mtile0 / mtile1 accumulators
#pragma unroll
    for (int i = 0; i < 8; ++i) {
        v8f zero = {0.f, 0.f, 0.f, 0.f, 0.f, 0.f, 0.f, 0.f};
        acc0[i] = zero;
        acc1[i] = zero;
    }

    const uint4* bbase  = reinterpret_cast<const uint4*>(bswz);
    const char*  bbytes = reinterpret_cast<const char*>(bswz);

    for (int kt = 0; kt < 32; ++kt) {
        // Prefetch next K-step's B fragments (8 KB slab, L2-resident) into
        // the near cache while this step's WMMAs run. Zero VGPR cost.
        if (kt + 1 < 32) {
#pragma unroll
            for (int nt = 0; nt < 8; ++nt) {
                size_t off = ((size_t)((wave * 8 + nt) * 32 + (kt + 1)) << 10)
                           + ((size_t)lane << 5);
                __builtin_prefetch(bbytes + off, 0, 3);
            }
        }

        // A fragments from LDS (ISA 16x32 bf16 layout):
        // halves 0-7 = K kk..kk+7, halves 8-15 = K kk+16..kk+23,
        // kk = kt*32 (+8 for lanes 16-31).
        const uint4* ap0 = reinterpret_cast<const uint4*>(smemA + mrow * HIDDEN + kt * 32 + hi);
        const uint4* ap1 = reinterpret_cast<const uint4*>(smemA + (16 + mrow) * HIDDEN + kt * 32 + hi);
        v16bf a0 = packbf(ap0[0], ap0[2]);
        v16bf a1 = packbf(ap1[0], ap1[2]);

#pragma unroll
        for (int nt = 0; nt < 8; ++nt) {
            // B fragment: 1024B block = 64 uint4; lane's 32B at lane*2 uint4s.
            const uint4* bp = bbase +
                ((size_t)((wave * 8 + nt) * 32 + kt) * 64 + lane * 2);
            v16bf bfrag = packbf(bp[0], bp[1]);

            acc0[nt] = __builtin_amdgcn_wmma_f32_16x16x32_bf16(
                false, a0, false, bfrag, (short)0, acc0[nt], false, false);
            acc1[nt] = __builtin_amdgcn_wmma_f32_16x16x32_bf16(
                false, a1, false, bfrag, (short)0, acc1[nt], false, false);
        }
    }

    // ---- store: C/D layout: VGPR r -> row r (lanes 0-15) / row 8+r (lanes 16-31)
    const int row0 = mblk * 32 + hi;
#pragma unroll
    for (int nt = 0; nt < 8; ++nt) {
        int col = nbase + nt * 16 + mrow;
#pragma unroll
        for (int r = 0; r < 8; ++r) {
            z[(size_t)(row0 + r)      * FEATURES + col] = acc0[nt][r];
            z[(size_t)(row0 + 16 + r) * FEATURES + col] = acc1[nt][r];
        }
    }
}

// -------------------------------------------------------------------------
extern "C" void kernel_launch(void* const* d_in, const int* in_sizes, int n_in,
                              void* d_out, int out_size, void* d_ws, size_t ws_size,
                              hipStream_t stream) {
    (void)in_sizes; (void)n_in; (void)out_size; (void)ws_size;

    const float* inputs  = (const float*)d_in[0];   // [4096,8,1024] f32
    const float* scale   = (const float*)d_in[1];   // [1024] f32
    const float* wkernel = (const float*)d_in[2];   // [1024,1024] f32

    float* z  = (float*)d_out;                      // first output  [32768,1024]
    float* ln = (float*)d_out + Z_ELEMS;            // second output [32768,1024]
    unsigned short* bswz = (unsigned short*)d_ws;   // 2 MB bf16 swizzled weights

    // 1) weights -> bf16 WMMA-B fragment layout (1M elements)
    prep_b_kernel<<<(HIDDEN * FEATURES) / 256, 256, 0, stream>>>(wkernel, bswz);
    // 2) LayerNorm: one block per row
    ln_kernel<<<ROWS, 256, 0, stream>>>(inputs, scale, ln);
    // 3) WMMA GEMM: 1024 blocks x (8 waves, 128 cols x 32 rows each)
    gemm_kernel<<<ROWS / 32, 256, 0, stream>>>(ln, bswz, z);
}
